// Model_MoE_FBCSP_64106681860774
// MI455X (gfx1250) — matrix operand order repaired and checked by reading, not compiled, not run on hardware
//
#include <hip/hip_runtime.h>
#include <hip/hip_bf16.h>

// ---------------------------------------------------------------------------
// Model dims (from reference)
// ---------------------------------------------------------------------------
#define kB   16
#define kT   512
#define kD   512
#define kH   8
#define kDH  64
#define kL   6
#define kE   8
#define kF   2048
#define kNB  55
#define kNC  8
#define kIN  440
#define kINP 448     // K padded to multiple of 32
#define kNTOK 8192   // B*T
#define kC   2
#define kEPS 1e-5f

typedef unsigned int u32;
typedef unsigned short u16;
typedef __attribute__((ext_vector_type(4)))  u32    u32x4;
typedef __attribute__((ext_vector_type(4)))  int    i32x4;
typedef __attribute__((ext_vector_type(16))) __bf16 bf16x16;
typedef __attribute__((ext_vector_type(8)))  float  f32x8;

union FragU { bf16x16 f; u32x4 u[2]; };

// gfx1250 async global->LDS path (ASYNCcnt), guarded so we fall back to the
// sync path if this toolchain doesn't declare the builtins.
// Prototype (from hipcc diagnostics): (int4 AS1* src, int4 AS3* dst, imm, imm)
#if __has_builtin(__builtin_amdgcn_global_load_async_to_lds_b128) && \
    __has_builtin(__builtin_amdgcn_s_wait_asynccnt)
#define USE_ASYNC_LDS 1
typedef __attribute__((address_space(1))) i32x4* gasync_p;
typedef __attribute__((address_space(3))) i32x4* lasync_p;
#else
#define USE_ASYNC_LDS 0
#endif

__device__ __forceinline__ u16 f2bf(float f) {
  u32 x = __float_as_uint(f);
  u32 r = x + 0x7FFFu + ((x >> 16) & 1u);   // RNE
  return (u16)(r >> 16);
}
__device__ __forceinline__ float bf2f(u16 u) {
  return __uint_as_float(((u32)u) << 16);
}
__device__ __forceinline__ float gelu_tanh(float x) {
  float x3 = x * x * x;
  float t = tanhf(0.7978845608028654f * (x + 0.044715f * x3));
  return 0.5f * x * (1.0f + t);
}

// ---------------------------------------------------------------------------
// Generic WMMA GEMM:  C[M,N] = epilogue( A[M,K](bf16) @ Bt[N,K](bf16)^T )
// Block tile 256x64, K-step 32, 8 waves (wave32) in a 4(M)x2(N) grid; each
// wave owns a 64x32 sub-tile = 4x2 v_wmma_f32_16x16x32_bf16 per K-step
// (8 WMMA per 12 LDS fragment loads).
// Requires: M % 256 == 0, N % 64 == 0, K % 32 == 0 (true for every call site).
// flags: 1=gelu, 2=accumulate(f32 C), 4=bf16 out, 8=transposed-V scatter
// ---------------------------------------------------------------------------
#define TM 256
#define TN 64
#define TK 32
#define LDSS 40   // LDS row stride in elements (80B, 16B aligned, padded)

__global__ __launch_bounds__(256)
void k_gemm(const u16* __restrict__ A, const u16* __restrict__ Bt,
            void* __restrict__ Cout,
            const float* __restrict__ bias,
            const float* __restrict__ rowscale, int rs_stride,
            float alpha, int flags,
            int M, int N, int K,
            int ldA, int ldBt, int ldC,
            int zH,
            long long sAb, long long sAh,
            long long sBb, long long sBh,
            long long sCb, long long sCh,
            int DHd, int Td)
{
  __shared__ __align__(16) u16 sA[TM][LDSS];
  __shared__ __align__(16) u16 sB[TN][LDSS];

  const int tid  = threadIdx.x;
  const int lane = tid & 31;
  const int wid  = tid >> 5;
  const int wm   = (wid & 3) * 64;   // wave M offset within block tile
  const int wn   = (wid >> 2) * 32;  // wave N offset within block tile
  const int bm   = blockIdx.y * TM;
  const int bn   = blockIdx.x * TN;
  const int z    = blockIdx.z;
  const int zb   = z / zH;
  const int zh   = z - zb * zH;

  const u16* Ab = A + (long long)zb * sAb + (long long)zh * sAh;
  const u16* Bb = Bt + (long long)zb * sBb + (long long)zh * sBh;
  const long long cb = (long long)zb * sCb + (long long)zh * sCh;

  f32x8 acc[4][2];
#pragma unroll
  for (int i = 0; i < 4; ++i)
#pragma unroll
    for (int j = 0; j < 2; ++j)
#pragma unroll
      for (int r = 0; r < 8; ++r) acc[i][j][r] = 0.0f;

  const int arow = tid;               // 0..255, one A-tile row per thread
  const int brow = tid >> 2;          // 0..63
  const int bkz  = (tid & 3) * 8;     // 0,8,16,24
  const int fr   = lane & 15;
  const int kg   = lane >> 4;

  for (int k0 = 0; k0 < K; k0 += TK) {
    // ---- cooperative tile staging (global -> LDS) ----
    const u16* ga = Ab + (size_t)(bm + arow) * ldA + k0;
    const u16* gb = Bb + (size_t)(bn + brow) * ldBt + k0 + bkz;
#if USE_ASYNC_LDS
    // CDNA5 direct global->LDS async copy, 16B per lane per issue.
#pragma unroll
    for (int c = 0; c < 4; ++c)
      __builtin_amdgcn_global_load_async_to_lds_b128(
          (gasync_p)(ga + c * 8), (lasync_p)(&sA[arow][c * 8]), 0, 0);
    __builtin_amdgcn_global_load_async_to_lds_b128(
        (gasync_p)gb, (lasync_p)(&sB[brow][bkz]), 0, 0);
    __builtin_amdgcn_s_wait_asynccnt(0);
#else
    const u32x4* ag = (const u32x4*)ga;
#pragma unroll
    for (int c = 0; c < 4; ++c) *(u32x4*)&sA[arow][c * 8] = ag[c];
    *(u32x4*)&sB[brow][bkz] = *(const u32x4*)gb;
#endif
    if (k0 + TK < K) {   // prefetch next K tile -> global_prefetch_b8
      __builtin_prefetch(ga + TK, 0, 1);
      __builtin_prefetch(gb + TK, 0, 1);
    }
    __syncthreads();

    // ---- fragments from LDS per documented 16-bit WMMA layouts ----
    FragU af[4], bfr[2];
#pragma unroll
    for (int i = 0; i < 4; ++i) {
      const u16* p = &sA[wm + i * 16 + fr][kg * 8];
      af[i].u[0] = *(const u32x4*)p;        // K = kg*8 .. +7
      af[i].u[1] = *(const u32x4*)(p + 16); // K = kg*8+16 .. +23
    }
#pragma unroll
    for (int j = 0; j < 2; ++j) {
      const u16* p = &sB[wn + j * 16 + fr][kg * 8];
      bfr[j].u[0] = *(const u32x4*)p;        // K = kg*8 .. +7
      bfr[j].u[1] = *(const u32x4*)(p + 16); // K = kg*8+16 .. +23
    }
#pragma unroll
    for (int i = 0; i < 4; ++i)
#pragma unroll
      for (int j = 0; j < 2; ++j)
        acc[i][j] = __builtin_amdgcn_wmma_f32_16x16x32_bf16(
            false, af[i].f, false, bfr[j].f, (short)0, acc[i][j], false, false);
    __syncthreads();
  }

  // ---- epilogue ----
  const int lm = (lane >> 4) * 8;
  const int ln = lane & 15;
  const int cgelu = flags & 1, cacc = flags & 2, cobf = flags & 4, cvt = flags & 8;
#pragma unroll
  for (int i = 0; i < 4; ++i) {
#pragma unroll
    for (int j = 0; j < 2; ++j) {
#pragma unroll
      for (int r = 0; r < 8; ++r) {
        int m = bm + wm + i * 16 + lm + r;
        int n = bn + wn + j * 16 + ln;
        if (m >= M || n >= N) continue;
        float v = acc[i][j][r] * alpha;
        if (bias) v += bias[n];
        if (cgelu) v = gelu_tanh(v);
        if (rowscale) v *= rowscale[(size_t)m * rs_stride];
        long long cidx;
        if (cvt) {  // store V transposed: [b][h][dh][t]
          int hh = n / DHd, dd = n - hh * DHd;
          cidx = cb + (long long)hh * DHd * Td + (long long)dd * Td + m;
        } else {
          cidx = cb + (long long)m * ldC + n;
        }
        if (cobf) {
          ((u16*)Cout)[cidx] = f2bf(v);
        } else {
          float* Cf = (float*)Cout;
          float o = v;
          if (cacc) o += Cf[cidx];
          Cf[cidx] = o;
        }
      }
    }
  }
}

// ---------------------------------------------------------------------------
// Weight transpose + fp32 -> bf16 (+K padding):  dst[n*Kpad+k] = src[k*N+n]
// ---------------------------------------------------------------------------
__global__ __launch_bounds__(256)
void k_transpose(const float* __restrict__ src, u16* __restrict__ dst,
                 int K, int N, int Kpad, long long sstride, long long dstride)
{
  long long idx = (long long)blockIdx.x * 256 + threadIdx.x;
  long long per = (long long)N * Kpad;
  if (idx >= per) return;
  int n = (int)(idx / Kpad);
  int k = (int)(idx - (long long)n * Kpad);
  long long zi = blockIdx.z;
  dst[zi * dstride + idx] =
      (k < K) ? f2bf(src[zi * sstride + (size_t)k * N + n]) : (u16)0;
}

// x[b, nb, t, nc] -> A0[(b*T+t), nb*NC+nc] (bf16, K padded to 448)
__global__ __launch_bounds__(256)
void k_permute_x(const float* __restrict__ x, u16* __restrict__ A0)
{
  long long idx = (long long)blockIdx.x * 256 + threadIdx.x;
  if (idx >= (long long)kNTOK * kINP) return;
  int row = (int)(idx / kINP);
  int c   = (int)(idx - (long long)row * kINP);
  u16 v = 0;
  if (c < kIN) {
    int b = row / kT, t = row - b * kT;
    int nb = c / kNC, nc = c - nb * kNC;
    v = f2bf(x[(((size_t)b * kNB + nb) * kT + t) * kNC + nc]);
  }
  A0[idx] = v;
}

__global__ __launch_bounds__(256)
void k_zero(float* p, int n)
{
  int i = blockIdx.x * 256 + threadIdx.x;
  if (i < n) p[i] = 0.0f;
}

// ---------------------------------------------------------------------------
// LayerNorm, one wave32 per 512-element row. Optional additive input (pos),
// optional f32 and/or bf16 outputs.
// ---------------------------------------------------------------------------
__global__ __launch_bounds__(256)
void k_ln(const float* __restrict__ in, const float* __restrict__ addv, int addMod,
          const float* __restrict__ sc, const float* __restrict__ bi,
          float* __restrict__ outf, u16* __restrict__ outbf, int R)
{
  int wid = threadIdx.x >> 5, lane = threadIdx.x & 31;
  int r = blockIdx.x * 8 + wid;
  if (r >= R) return;
  const float* row = in + (size_t)r * kD;
  float v[16];
#pragma unroll
  for (int i = 0; i < 16; ++i) {
    int d = i * 32 + lane;
    float x = row[d];
    if (addv) x += addv[(size_t)(addMod ? (r % addMod) : r) * kD + d];
    v[i] = x;
  }
  float s = 0.0f;
#pragma unroll
  for (int i = 0; i < 16; ++i) s += v[i];
#pragma unroll
  for (int off = 16; off > 0; off >>= 1) s += __shfl_down(s, off, 32);
  s = __shfl(s, 0, 32);
  float mean = s * (1.0f / kD);
  float var = 0.0f;
#pragma unroll
  for (int i = 0; i < 16; ++i) { float dlt = v[i] - mean; var += dlt * dlt; }
#pragma unroll
  for (int off = 16; off > 0; off >>= 1) var += __shfl_down(var, off, 32);
  var = __shfl(var, 0, 32);
  float inv = rsqrtf(var * (1.0f / kD) + kEPS);
#pragma unroll
  for (int i = 0; i < 16; ++i) {
    int d = i * 32 + lane;
    float y = (v[i] - mean) * inv * sc[d] + bi[d];
    if (outf)  outf[(size_t)r * kD + d] = y;
    if (outbf) outbf[(size_t)r * kD + d] = f2bf(y);
  }
}

// Row softmax (512 wide), fp32 in -> bf16 out. One wave32 per row.
__global__ __launch_bounds__(256)
void k_softmax(const float* __restrict__ S, u16* __restrict__ att, long long rows)
{
  int wid = threadIdx.x >> 5, lane = threadIdx.x & 31;
  long long r = (long long)blockIdx.x * 8 + wid;
  if (r >= rows) return;
  const float* row = S + r * kT;
  float v[16];
  float mx = -3.4e38f;
#pragma unroll
  for (int i = 0; i < 16; ++i) { v[i] = row[i * 32 + lane]; mx = fmaxf(mx, v[i]); }
#pragma unroll
  for (int off = 16; off > 0; off >>= 1) mx = fmaxf(mx, __shfl_down(mx, off, 32));
  mx = __shfl(mx, 0, 32);
  float s = 0.0f;
#pragma unroll
  for (int i = 0; i < 16; ++i) { v[i] = expf(v[i] - mx); s += v[i]; }
#pragma unroll
  for (int off = 16; off > 0; off >>= 1) s += __shfl_down(s, off, 32);
  s = __shfl(s, 0, 32);
  float invs = 1.0f / s;
#pragma unroll
  for (int i = 0; i < 16; ++i) att[r * kT + i * 32 + lane] = f2bf(v[i] * invs);
}

// ---------------------------------------------------------------------------
// MoE gate: softmax over 8 experts, top-2 renormalized combine weights,
// aux-loss statistics (routing counts + mean probs) via atomics.
// One wave32 per token.
// ---------------------------------------------------------------------------
__global__ __launch_bounds__(256)
void k_gate(const u16* __restrict__ mbf, const float* __restrict__ gw,
            float* __restrict__ combine, float* __restrict__ auxbuf)
{
  int wid = threadIdx.x >> 5, lane = threadIdx.x & 31;
  int t = blockIdx.x * 8 + wid;
  if (t >= kNTOK) return;
  float acc[kE];
#pragma unroll
  for (int e = 0; e < kE; ++e) acc[e] = 0.0f;
  for (int d = lane; d < kD; d += 32) {
    float x = bf2f(mbf[(size_t)t * kD + d]);
#pragma unroll
    for (int e = 0; e < kE; ++e) acc[e] += x * gw[d * kE + e];
  }
#pragma unroll
  for (int e = 0; e < kE; ++e) {
#pragma unroll
    for (int off = 16; off > 0; off >>= 1) acc[e] += __shfl_down(acc[e], off, 32);
  }
  if (lane == 0) {
    float mx = acc[0];
#pragma unroll
    for (int e = 1; e < kE; ++e) mx = fmaxf(mx, acc[e]);
    float p[kE], s = 0.0f;
#pragma unroll
    for (int e = 0; e < kE; ++e) { p[e] = expf(acc[e] - mx); s += p[e]; }
    float invs = 1.0f / s;
#pragma unroll
    for (int e = 0; e < kE; ++e) p[e] *= invs;
    int i1 = 0;
#pragma unroll
    for (int e = 1; e < kE; ++e) if (p[e] > p[i1]) i1 = e;   // first max (jax tie rule)
    int i2 = (i1 == 0) ? 1 : 0;
#pragma unroll
    for (int e = 0; e < kE; ++e) if (e != i1 && e != i2 && p[e] > p[i2]) i2 = e;
    float wsum = p[i1] + p[i2];
#pragma unroll
    for (int e = 0; e < kE; ++e) combine[(size_t)t * kE + e] = 0.0f;
    combine[(size_t)t * kE + i1] = p[i1] / wsum;
    combine[(size_t)t * kE + i2] = p[i2] / wsum;
    atomicAdd(&auxbuf[i1], 1.0f);
    atomicAdd(&auxbuf[i2], 1.0f);
#pragma unroll
    for (int e = 0; e < kE; ++e) atomicAdd(&auxbuf[kE + e], p[e]);
  }
}

// ---------------------------------------------------------------------------
// Head: mean-pool over T, final LN, logits = @head_w + head_b, aux loss.
// One block per batch element.
// ---------------------------------------------------------------------------
__global__ __launch_bounds__(256)
void k_head(const float* __restrict__ h, const float* __restrict__ lnfs,
            const float* __restrict__ lnfb, const float* __restrict__ hw,
            const float* __restrict__ hb, const float* __restrict__ auxbuf,
            float* __restrict__ out)
{
  __shared__ float pooled[kD];
  __shared__ float stats[2];
  int b = blockIdx.x;
  for (int d = threadIdx.x; d < kD; d += blockDim.x) {
    float s = 0.0f;
    for (int t = 0; t < kT; ++t) s += h[((size_t)(b * kT + t)) * kD + d];
    pooled[d] = s * (1.0f / kT);
  }
  __syncthreads();
  if (threadIdx.x == 0) {
    float m = 0.0f;
    for (int d = 0; d < kD; ++d) m += pooled[d];
    m *= (1.0f / kD);
    float v = 0.0f;
    for (int d = 0; d < kD; ++d) { float dl = pooled[d] - m; v += dl * dl; }
    stats[0] = m;
    stats[1] = rsqrtf(v * (1.0f / kD) + kEPS);
  }
  __syncthreads();
  if (threadIdx.x < kC) {
    int c = threadIdx.x;
    float acc = hb[c];
    for (int d = 0; d < kD; ++d) {
      float y = (pooled[d] - stats[0]) * stats[1] * lnfs[d] + lnfb[d];
      acc += y * hw[d * kC + c];
    }
    out[b * kC + c] = acc;
  }
  if (b == 0 && threadIdx.x == 0) {
    float total = 0.0f;
    for (int l = 0; l < kL; ++l) {
      float a = 0.0f;
      for (int e = 0; e < kE; ++e)
        a += (auxbuf[l * 16 + e] * (1.0f / kNTOK)) *
             (auxbuf[l * 16 + kE + e] * (1.0f / kNTOK));
      total += (float)kE * a;
    }
    out[kB * kC] = total;
  }
}

// ---------------------------------------------------------------------------
// Host side
// ---------------------------------------------------------------------------
static void launch_gemm(hipStream_t s, const u16* A, const u16* Bt, void* C,
                        const float* bias, const float* rowscale, int rs_stride,
                        float alpha, int flags, int M, int N, int K,
                        int ldA, int ldBt, int ldC, int Z, int zH,
                        long long sAb, long long sAh, long long sBb, long long sBh,
                        long long sCb, long long sCh, int DHd, int Td)
{
  dim3 g(N / TN, M / TM, Z);
  k_gemm<<<g, 256, 0, s>>>(A, Bt, C, bias, rowscale, rs_stride, alpha, flags,
                           M, N, K, ldA, ldBt, ldC, zH,
                           sAb, sAh, sBb, sBh, sCb, sCh, DHd, Td);
}

extern "C" void kernel_launch(void* const* d_in, const int* in_sizes, int n_in,
                              void* d_out, int out_size, void* d_ws, size_t ws_size,
                              hipStream_t stream)
{
  const float* x        = (const float*)d_in[0];
  const float* proj_w   = (const float*)d_in[1];
  const float* proj_b   = (const float*)d_in[2];
  const float* pos      = (const float*)d_in[3];
  const float* ln_pre_s = (const float*)d_in[4];
  const float* ln_pre_b = (const float*)d_in[5];
  const float* ln1_s    = (const float*)d_in[6];
  const float* ln1_b    = (const float*)d_in[7];
  const float* wq       = (const float*)d_in[8];
  const float* bq       = (const float*)d_in[9];
  const float* wk       = (const float*)d_in[10];
  const float* bk       = (const float*)d_in[11];
  const float* wv       = (const float*)d_in[12];
  const float* bv       = (const float*)d_in[13];
  const float* wo       = (const float*)d_in[14];
  const float* bo       = (const float*)d_in[15];
  const float* ln2_s    = (const float*)d_in[16];
  const float* ln2_b    = (const float*)d_in[17];
  const float* gate_w   = (const float*)d_in[18];
  const float* e_w1     = (const float*)d_in[19];
  const float* e_b1     = (const float*)d_in[20];
  const float* e_w2     = (const float*)d_in[21];
  const float* e_b2     = (const float*)d_in[22];
  const float* ln_f_s   = (const float*)d_in[23];
  const float* ln_f_b   = (const float*)d_in[24];
  const float* head_w   = (const float*)d_in[25];
  const float* head_b   = (const float*)d_in[26];
  (void)in_sizes; (void)n_in; (void)out_size; (void)ws_size;

  // ---- bump allocator over d_ws (total ~524 MB) ----
  char* p = (char*)d_ws;
  auto alloc = [&](size_t bytes) -> void* {
    void* r = (void*)p;
    p += (bytes + 255) & ~(size_t)255;
    return r;
  };
  u16*   A0    = (u16*)alloc((size_t)kNTOK * kINP * 2);
  u16*   projT = (u16*)alloc((size_t)kD * kINP * 2);
  u16*   wqT   = (u16*)alloc((size_t)kL * kD * kD * 2);
  u16*   wkT   = (u16*)alloc((size_t)kL * kD * kD * 2);
  u16*   wvT   = (u16*)alloc((size_t)kL * kD * kD * 2);
  u16*   woT   = (u16*)alloc((size_t)kL * kD * kD * 2);
  u16*   w1T   = (u16*)alloc((size_t)kL * kE * kF * kD * 2);
  u16*   w2T   = (u16*)alloc((size_t)kL * kE * kD * kF * 2);
  float* h     = (float*)alloc((size_t)kNTOK * kD * 4);
  u16*   a_bf  = (u16*)alloc((size_t)kNTOK * kD * 2);
  u16*   qb    = (u16*)alloc((size_t)kNTOK * kD * 2);
  u16*   kb    = (u16*)alloc((size_t)kNTOK * kD * 2);
  u16*   vt    = (u16*)alloc((size_t)kB * kH * kDH * kT * 2);
  float* scores= (float*)alloc((size_t)kB * kH * kT * kT * 4);
  u16*   att   = (u16*)alloc((size_t)kB * kH * kT * kT * 2);
  u16*   ctx   = (u16*)alloc((size_t)kNTOK * kD * 2);
  u16*   m_bf  = (u16*)alloc((size_t)kNTOK * kD * 2);
  u16*   g_bf  = (u16*)alloc((size_t)kNTOK * kF * 2);
  float* comb  = (float*)alloc((size_t)kNTOK * kE * 4);
  float* auxb  = (float*)alloc((size_t)kL * 16 * 4);

  // ---- one-time (per launch) weight conversion/transposition to bf16 ----
  {
    long long per = (long long)kD * kINP;                 // projT
    k_transpose<<<dim3((unsigned)((per + 255) / 256), 1, 1), 256, 0, stream>>>(
        proj_w, projT, kIN, kD, kINP, 0, 0);
  }
  {
    long long per = (long long)kD * kD;
    dim3 g((unsigned)((per + 255) / 256), 1, kL);
    k_transpose<<<g, 256, 0, stream>>>(wq, wqT, kD, kD, kD, per, per);
    k_transpose<<<g, 256, 0, stream>>>(wk, wkT, kD, kD, kD, per, per);
    k_transpose<<<g, 256, 0, stream>>>(wv, wvT, kD, kD, kD, per, per);
    k_transpose<<<g, 256, 0, stream>>>(wo, woT, kD, kD, kD, per, per);
  }
  {
    long long per = (long long)kF * kD;                   // w1: [D,F] -> [F,D]
    dim3 g((unsigned)((per + 255) / 256), 1, kL * kE);
    k_transpose<<<g, 256, 0, stream>>>(e_w1, w1T, kD, kF, kD, per, per);
  }
  {
    long long per = (long long)kD * kF;                   // w2: [F,D] -> [D,F]
    dim3 g((unsigned)((per + 255) / 256), 1, kL * kE);
    k_transpose<<<g, 256, 0, stream>>>(e_w2, w2T, kF, kD, kF, per, per);
  }
  k_zero<<<1, 128, 0, stream>>>(auxb, kL * 16);

  // ---- frontend: permute x, project, +pos, pre-LN ----
  {
    long long tot = (long long)kNTOK * kINP;
    k_permute_x<<<(unsigned)((tot + 255) / 256), 256, 0, stream>>>(x, A0);
  }
  launch_gemm(stream, A0, projT, h, proj_b, nullptr, 0, 1.0f, /*flags*/0,
              kNTOK, kD, kINP, kINP, kINP, kD, 1, 1, 0, 0, 0, 0, 0, 0, kDH, kT);
  k_ln<<<kNTOK / 8, 256, 0, stream>>>(h, pos, kT, ln_pre_s, ln_pre_b, h, nullptr, kNTOK);

  const float attn_scale = 1.0f / 8.0f;  // DH^-0.5
  for (int l = 0; l < kL; ++l) {
    const u16* wqTl = wqT + (size_t)l * kD * kD;
    const u16* wkTl = wkT + (size_t)l * kD * kD;
    const u16* wvTl = wvT + (size_t)l * kD * kD;
    const u16* woTl = woT + (size_t)l * kD * kD;

    // a = LN1(h)  (bf16)
    k_ln<<<kNTOK / 8, 256, 0, stream>>>(h, nullptr, 0, ln1_s + l * kD, ln1_b + l * kD,
                                        nullptr, a_bf, kNTOK);
    // q, k  (bf16, [B*T, D])
    launch_gemm(stream, a_bf, wqTl, qb, bq + l * kD, nullptr, 0, 1.0f, /*bf16*/4,
                kNTOK, kD, kD, kD, kD, kD, 1, 1, 0, 0, 0, 0, 0, 0, kDH, kT);
    launch_gemm(stream, a_bf, wkTl, kb, bk + l * kD, nullptr, 0, 1.0f, 4,
                kNTOK, kD, kD, kD, kD, kD, 1, 1, 0, 0, 0, 0, 0, 0, kDH, kT);
    // v, stored transposed as vt[b][h][dh][t]  (batched over b)
    launch_gemm(stream, a_bf, wvTl, vt, bv + l * kD, nullptr, 0, 1.0f, /*bf16|vt*/4 | 8,
                kT, kD, kD, kD, kD, kD, kB, 1,
                (long long)kT * kD, 0, 0, 0, (long long)kH * kDH * kT, 0, kDH, kT);
    // scores[b,h] = q[b,h] @ k[b,h]^T * scale   (Z = B*H)
    launch_gemm(stream, qb, kb, scores, nullptr, nullptr, 0, attn_scale, 0,
                kT, kT, kDH, kD, kD, kT, kB * kH, kH,
                (long long)kT * kD, kDH, (long long)kT * kD, kDH,
                (long long)kH * kT * kT, (long long)kT * kT, kDH, kT);
    // softmax rows -> bf16 att
    k_softmax<<<(kB * kH * kT) / 8, 256, 0, stream>>>(scores, att, (long long)kB * kH * kT);
    // ctx[b,h] = att[b,h] @ v[b,h]   (Bt = vt[b][h] is [DH,T], contiguous)
    launch_gemm(stream, att, vt, ctx, nullptr, nullptr, 0, 1.0f, 4,
                kT, kDH, kT, kT, kT, kD, kB * kH, kH,
                (long long)kH * kT * kT, (long long)kT * kT,
                (long long)kH * kDH * kT, (long long)kDH * kT,
                (long long)kT * kD, kDH, kDH, kT);
    // h += ctx @ wo + bo   (residual)
    launch_gemm(stream, ctx, woTl, h, bo + l * kD, nullptr, 0, 1.0f, /*acc*/2,
                kNTOK, kD, kD, kD, kD, kD, 1, 1, 0, 0, 0, 0, 0, 0, kDH, kT);

    // m = LN2(h) (bf16); gate -> combine + aux stats
    k_ln<<<kNTOK / 8, 256, 0, stream>>>(h, nullptr, 0, ln2_s + l * kD, ln2_b + l * kD,
                                        nullptr, m_bf, kNTOK);
    k_gate<<<kNTOK / 8, 256, 0, stream>>>(m_bf, gate_w + (size_t)l * kD * kE,
                                          comb, auxb + l * 16);
    // experts (dense evaluation; combine==0 zeroes non-selected contributions)
    for (int e = 0; e < kE; ++e) {
      const u16* w1e = w1T + ((size_t)l * kE + e) * kF * kD;
      const u16* w2e = w2T + ((size_t)l * kE + e) * kD * kF;
      const float* b1e = e_b1 + ((size_t)l * kE + e) * kF;
      const float* b2e = e_b2 + ((size_t)l * kE + e) * kD;
      // g = gelu(m @ w1 + b1)  (bf16)
      launch_gemm(stream, m_bf, w1e, g_bf, b1e, nullptr, 0, 1.0f, /*gelu|bf16*/1 | 4,
                  kNTOK, kF, kD, kD, kD, kF, 1, 1, 0, 0, 0, 0, 0, 0, kDH, kT);
      // h += combine[:,e] * (g @ w2 + b2)
      launch_gemm(stream, g_bf, w2e, h, b2e, comb + e, kE, 1.0f, /*acc*/2,
                  kNTOK, kD, kF, kF, kF, kD, 1, 1, 0, 0, 0, 0, 0, 0, kDH, kT);
    }
  }

  // pooled mean, final LN, head, aux
  k_head<<<kB, 256, 0, stream>>>(h, ln_f_s, ln_f_b, head_w, head_b, auxb, (float*)d_out);
}
